// GNNStackSolution_12601434046987
// MI455X (gfx1250) — matrix operand order repaired and checked by reading, not compiled, not run on hardware
//
#include <hip/hip_runtime.h>
#include <hip/hip_bf16.h>
#include <math.h>

typedef __attribute__((ext_vector_type(16))) _Float16 v16h;
typedef __attribute__((ext_vector_type(8)))  _Float16 v8h;
typedef __attribute__((ext_vector_type(4)))  _Float16 v4h;
typedef __attribute__((ext_vector_type(2)))  _Float16 v2h;
typedef __attribute__((ext_vector_type(8)))  float    v8f;

#define NEG_SLOPE 0.2f

// scalar zero-select (lowered to v_cndmask; aggregate ternaries would spill)
#define SEL(ok, x) ((ok) ? (x) : 0.0f)

// ---------------------------------------------------------------------------
// WMMA GEMM:  Y[M,Ncols] = X[M,K] * W[K,Ncols]   (f32 in/out, f16 multiply,
// f32 accumulate via v_wmma_f32_16x16x32_f16).
// Block = 256 threads = 8 wave32s, arranged 2(m) x 4(n).
// Block tile 64(M) x 128(N), K-step 32. Each wave computes a 32x32 tile:
// 2 A fragments + 2 B fragments -> 4 WMMAs per K-step.
// K must be a multiple of 32 (holds: 128 / 256 / 256).
// ---------------------------------------------------------------------------
__launch_bounds__(256)
__global__ void k_gemm_wmma(const float* __restrict__ X,
                            const float* __restrict__ Wm,
                            float* __restrict__ Y,
                            int M, int K, int Ncols)
{
    // rows padded to 40 halves (80 B): keeps 16B/8B/4B alignment for packed
    // LDS accesses and staggers banks.
    __shared__ _Float16 As[64][40];    // [m][k]
    __shared__ _Float16 Bs[128][40];   // [n][k]  (B transposed)

    const int tid  = threadIdx.x;
    const int lane = tid & 31;
    const int wave = tid >> 5;
    const int wm   = wave >> 2;              // 0..1 : 32-row slab
    const int wn   = wave & 3;               // 0..3 : 32-col slab
    const int g    = lane >> 4;              // lane group 0/1
    const int r    = lane & 15;
    const int mBase = blockIdx.x * 64;
    const int nBase = blockIdx.y * 128;

    // A-tile staging: 64x32 f32 = 512 float4; 2 per thread.
    const int am0 = tid >> 3;                // rows 0..31 (j=0), +32 (j=1)
    const int ak0 = (tid & 7) * 4;
    // B-tile staging: 32(k) x 128(n) f32 = 512 micro-tiles of 2(k)x4(n);
    // 2 micro-tiles per thread.
    const int bk  = (tid & 15) * 2;          // 0..30
    const int bn0 = (tid >> 4) * 4;          // 0..60  (j=0); +64 (j=1)

    // branchless OOB handling: clamp addresses to a legal row, zero by select.
    const int gm0 = mBase + am0;
    const int gm1 = gm0 + 32;
    const bool okA0 = gm0 < M;
    const bool okA1 = gm1 < M;
    const size_t rowA0 = (size_t)(okA0 ? gm0 : M - 1) * K;
    const size_t rowA1 = (size_t)(okA1 ? gm1 : M - 1) * K;
    const int gn0 = nBase + bn0;
    const int gn1 = gn0 + 64;
    const bool okB0 = (gn0 + 3) < Ncols;     // Ncols % 4 == 0 -> exact guard
    const bool okB1 = (gn1 + 3) < Ncols;
    const int gn0c = okB0 ? gn0 : 0;
    const int gn1c = okB1 ? gn1 : 0;

    v8f acc00 = {}, acc01 = {}, acc10 = {}, acc11 = {};

    for (int kb = 0; kb < K; kb += 32) {
        // ---- stage A tile: unconditional loads, per-component zero-select --
        {
            float4 v0 = *(const float4*)(X + rowA0 + kb + ak0);
            float4 v1 = *(const float4*)(X + rowA1 + kb + ak0);
            v4h h0 = { (_Float16)SEL(okA0, v0.x), (_Float16)SEL(okA0, v0.y),
                       (_Float16)SEL(okA0, v0.z), (_Float16)SEL(okA0, v0.w) };
            v4h h1 = { (_Float16)SEL(okA1, v1.x), (_Float16)SEL(okA1, v1.y),
                       (_Float16)SEL(okA1, v1.z), (_Float16)SEL(okA1, v1.w) };
            *(v4h*)&As[am0][ak0]      = h0;
            *(v4h*)&As[am0 + 32][ak0] = h1;
        }
        // ---- stage B tile transposed: two 2x4 micro-tiles ----
        const int gk = kb + bk;
        {
            float4 r0 = *(const float4*)(Wm + (size_t)gk * Ncols + gn0c);
            float4 r1 = *(const float4*)(Wm + (size_t)(gk + 1) * Ncols + gn0c);
            v2h p0 = { (_Float16)SEL(okB0, r0.x), (_Float16)SEL(okB0, r1.x) };
            v2h p1 = { (_Float16)SEL(okB0, r0.y), (_Float16)SEL(okB0, r1.y) };
            v2h p2 = { (_Float16)SEL(okB0, r0.z), (_Float16)SEL(okB0, r1.z) };
            v2h p3 = { (_Float16)SEL(okB0, r0.w), (_Float16)SEL(okB0, r1.w) };
            *(v2h*)&Bs[bn0 + 0][bk] = p0;
            *(v2h*)&Bs[bn0 + 1][bk] = p1;
            *(v2h*)&Bs[bn0 + 2][bk] = p2;
            *(v2h*)&Bs[bn0 + 3][bk] = p3;
        }
        {
            float4 r0 = *(const float4*)(Wm + (size_t)gk * Ncols + gn1c);
            float4 r1 = *(const float4*)(Wm + (size_t)(gk + 1) * Ncols + gn1c);
            v2h q0 = { (_Float16)SEL(okB1, r0.x), (_Float16)SEL(okB1, r1.x) };
            v2h q1 = { (_Float16)SEL(okB1, r0.y), (_Float16)SEL(okB1, r1.y) };
            v2h q2 = { (_Float16)SEL(okB1, r0.z), (_Float16)SEL(okB1, r1.z) };
            v2h q3 = { (_Float16)SEL(okB1, r0.w), (_Float16)SEL(okB1, r1.w) };
            *(v2h*)&Bs[bn0 + 64][bk] = q0;
            *(v2h*)&Bs[bn0 + 65][bk] = q1;
            *(v2h*)&Bs[bn0 + 66][bk] = q2;
            *(v2h*)&Bs[bn0 + 67][bk] = q3;
        }
        // prefetch next K tile (global_prefetch_b8; speculative)
        if (kb + 32 < K) {
            __builtin_prefetch(X + rowA0 + kb + 32, 0, 1);
            __builtin_prefetch(&Wm[(size_t)(kb + 32) * Ncols + nBase], 0, 1);
        }
        __syncthreads();

        // ---- fragments per ISA VGPR layouts (CDNA5 ISA) ----
        //  A (16x32 f16): M = lane%16 ; elem i -> K = i+8g (i<8), 16+(i-8)+8g
        //  B (32x16 f16): N = lane%16 ; elem i -> K = i+8g (i<8), 16+(i-8)+8g
        const int arow0 = wm * 32 + r;
        const int arow1 = arow0 + 16;
        v8h a0_lo = *(const v8h*)&As[arow0][8 * g];
        v8h a0_hi = *(const v8h*)&As[arow0][16 + 8 * g];
        v8h a1_lo = *(const v8h*)&As[arow1][8 * g];
        v8h a1_hi = *(const v8h*)&As[arow1][16 + 8 * g];
        const int brow0 = wn * 32 + r;
        const int brow1 = brow0 + 16;
        v8h b0_lo = *(const v8h*)&Bs[brow0][8 * g];
        v8h b0_hi = *(const v8h*)&Bs[brow0][16 + 8 * g];
        v8h b1_lo = *(const v8h*)&Bs[brow1][8 * g];
        v8h b1_hi = *(const v8h*)&Bs[brow1][16 + 8 * g];

        v16h a0, a1, b0, b1;
#pragma unroll
        for (int i = 0; i < 8; ++i) {
            a0[i] = a0_lo[i]; a0[i + 8] = a0_hi[i];
            a1[i] = a1_lo[i]; a1[i + 8] = a1_hi[i];
            b0[i] = b0_lo[i]; b0[i + 8] = b0_hi[i];
            b1[i] = b1_lo[i]; b1[i + 8] = b1_hi[i];
        }

        acc00 = __builtin_amdgcn_wmma_f32_16x16x32_f16(
                    false, a0, false, b0, (short)0, acc00, false, false);
        acc01 = __builtin_amdgcn_wmma_f32_16x16x32_f16(
                    false, a0, false, b1, (short)0, acc01, false, false);
        acc10 = __builtin_amdgcn_wmma_f32_16x16x32_f16(
                    false, a1, false, b0, (short)0, acc10, false, false);
        acc11 = __builtin_amdgcn_wmma_f32_16x16x32_f16(
                    false, a1, false, b1, (short)0, acc11, false, false);
        __syncthreads();
    }

    // C/D layout: elem rr -> row = rr + 8*g, col = lane%16
    const int colA = nBase + wn * 32 + r;
    const int colB = colA + 16;
    const int rowA = mBase + wm * 32 + 8 * g;
    const int rowB = rowA + 16;
#pragma unroll
    for (int rr = 0; rr < 8; ++rr) {
        int ra = rowA + rr;
        if (ra < M) {
            if (colA < Ncols) Y[(size_t)ra * Ncols + colA] = acc00[rr];
            if (colB < Ncols) Y[(size_t)ra * Ncols + colB] = acc01[rr];
        }
        int rb = rowB + rr;
        if (rb < M) {
            if (colA < Ncols) Y[(size_t)rb * Ncols + colA] = acc10[rr];
            if (colB < Ncols) Y[(size_t)rb * Ncols + colB] = acc11[rr];
        }
    }
}

// ---------------------------------------------------------------------------
// Per-node attention scores: s1[n,h]=h[n,h,:].a_src[h], s2 likewise a_dst.
// ---------------------------------------------------------------------------
template <int H, int C>
__global__ void k_scores(const float* __restrict__ h,
                         const float* __restrict__ a_src,
                         const float* __restrict__ a_dst,
                         float* __restrict__ s1, float* __restrict__ s2,
                         unsigned NH)
{
    unsigned t = blockIdx.x * blockDim.x + threadIdx.x;
    if (t >= NH) return;
    unsigned n = t / H, hh = t % H;
    const float* hp = h + ((size_t)n * H + hh) * C;
    const float* ap = a_src + (size_t)hh * C;
    const float* bp = a_dst + (size_t)hh * C;
    float x = 0.f, y = 0.f;
#pragma unroll
    for (int c = 0; c < C; ++c) { x += hp[c] * ap[c]; y += hp[c] * bp[c]; }
    s1[t] = x; s2[t] = y;
}

// monotone float<->uint key for atomic max over floats (incl. negatives)
__device__ __forceinline__ unsigned fkey(float f) {
    unsigned u = __float_as_uint(f);
    return (u & 0x80000000u) ? ~u : (u | 0x80000000u);
}
__device__ __forceinline__ float funkey(unsigned k) {
    unsigned u = (k & 0x80000000u) ? (k & 0x7FFFFFFFu) : ~k;
    return __uint_as_float(u);
}
#define KEY_NEG_INF 0x007FFFFFu   // fkey(-inf)

__global__ void k_fill_u32(unsigned* __restrict__ p, unsigned v, unsigned n) {
    unsigned t = blockIdx.x * blockDim.x + threadIdx.x;
    if (t < n) p[t] = v;
}
// vectorized fill: n4 = count/4 (count is a multiple of 4, 16B-aligned ptr)
__global__ void k_fill_f32x4(float4* __restrict__ p, float v, unsigned n4) {
    unsigned t = blockIdx.x * blockDim.x + threadIdx.x;
    if (t < n4) p[t] = make_float4(v, v, v, v);
}

// alpha = leakyrelu(s1[dst]+s2[src]); segment max by dst (atomic on key)
template <int H>
__global__ void k_edge_alpha(const int* __restrict__ src, const int* __restrict__ dst,
                             const float* __restrict__ s1, const float* __restrict__ s2,
                             float* __restrict__ alpha, unsigned* __restrict__ amax,
                             unsigned EH)
{
    unsigned t = blockIdx.x * blockDim.x + threadIdx.x;
    if (t >= EH) return;
    unsigned e = t / H, hh = t % H;
    unsigned d = (unsigned)dst[e], s = (unsigned)src[e];
    float a = s1[d * H + hh] + s2[s * H + hh];
    a = (a > 0.f) ? a : NEG_SLOPE * a;
    alpha[t] = a;
    atomicMax(&amax[d * H + hh], fkey(a));
}

// ex = exp(alpha - amax[dst]); segment sum by dst
template <int H>
__global__ void k_edge_exp(const int* __restrict__ dst,
                           float* __restrict__ alpha,
                           const unsigned* __restrict__ amax,
                           float* __restrict__ denom,
                           unsigned EH)
{
    unsigned t = blockIdx.x * blockDim.x + threadIdx.x;
    if (t >= EH) return;
    unsigned e = t / H, hh = t % H;
    unsigned d = (unsigned)dst[e];
    float ex = __expf(alpha[t] - funkey(amax[d * H + hh]));
    alpha[t] = ex;
    atomicAdd(&denom[d * H + hh], ex);
}

// coef = ex / denom[dst]  (folded in place so the element pass avoids divides)
template <int H>
__global__ void k_edge_coef(const int* __restrict__ dst,
                            float* __restrict__ coef,
                            const float* __restrict__ denom,
                            unsigned EH)
{
    unsigned t = blockIdx.x * blockDim.x + threadIdx.x;
    if (t >= EH) return;
    unsigned e = t / H, hh = t % H;
    unsigned d = (unsigned)dst[e];
    coef[t] = coef[t] / denom[d * H + hh];
}

// out[dst,h,c] += coef[e,h] * h[src,h,c] ; one thread per (e,h,c) so that
// consecutive lanes hit consecutive channels: coalesced gathers and
// coalesced global_atomic_add_f32 bursts on the destination cacheline.
template <int H, int C>
__global__ void k_edge_agg(const int* __restrict__ src, const int* __restrict__ dst,
                           const float* __restrict__ coef,
                           const float* __restrict__ h, float* __restrict__ out,
                           unsigned EHC)
{
    unsigned t = blockIdx.x * blockDim.x + threadIdx.x;
    if (t >= EHC) return;
    const unsigned HC = (unsigned)H * C;
    unsigned e  = t / HC;
    unsigned rc = t % HC;          // hh*C + c
    unsigned hh = rc / C;
    unsigned s = (unsigned)src[e], d = (unsigned)dst[e];
    float v = coef[e * H + hh] * h[(size_t)s * HC + rc];
    atomicAdd(&out[(size_t)d * HC + rc], v);
}

// vectorized ELU over a multiple-of-4, 16B-aligned buffer
__global__ void k_elu4(float4* __restrict__ x, unsigned n4) {
    unsigned t = blockIdx.x * blockDim.x + threadIdx.x;
    if (t >= n4) return;
    float4 v = x[t];
    v.x = (v.x > 0.f) ? v.x : (__expf(v.x) - 1.0f);
    v.y = (v.y > 0.f) ? v.y : (__expf(v.y) - 1.0f);
    v.z = (v.z > 0.f) ? v.z : (__expf(v.z) - 1.0f);
    v.w = (v.w > 0.f) ? v.w : (__expf(v.w) - 1.0f);
    x[t] = v;
}

template <int C>
__global__ void k_logsoftmax(const float* __restrict__ in, float* __restrict__ out,
                             unsigned N)
{
    unsigned t = blockIdx.x * blockDim.x + threadIdx.x;
    if (t >= N) return;
    const float* ip = in + (size_t)t * C;
    float m = -1e30f;
#pragma unroll
    for (int c = 0; c < C; ++c) m = fmaxf(m, ip[c]);
    float s = 0.f;
#pragma unroll
    for (int c = 0; c < C; ++c) s += __expf(ip[c] - m);
    float lse = m + __logf(s);
    float* op = out + (size_t)t * C;
#pragma unroll
    for (int c = 0; c < C; ++c) op[c] = ip[c] - lse;
}

// ---------------------------------------------------------------------------
static inline unsigned cdiv(unsigned a, unsigned b) { return (a + b - 1) / b; }

extern "C" void kernel_launch(void* const* d_in, const int* in_sizes, int n_in,
                              void* d_out, int out_size, void* d_ws, size_t ws_size,
                              hipStream_t stream)
{
    const float* x   = (const float*)d_in[0];
    const int*   ei  = (const int*)d_in[1];
    const float* W1  = (const float*)d_in[2];
    const float* as1 = (const float*)d_in[3];
    const float* ad1 = (const float*)d_in[4];
    const float* W2  = (const float*)d_in[5];
    const float* as2 = (const float*)d_in[6];
    const float* ad2 = (const float*)d_in[7];
    const float* W3  = (const float*)d_in[8];
    const float* as3 = (const float*)d_in[9];
    const float* ad3 = (const float*)d_in[10];

    const int IN = 128, H = 8, C = 32, HC = 256, OUT = 40;
    const int N = in_sizes[0] / IN;
    const int E = in_sizes[1] / 2;
    const int* src = ei;
    const int* dst = ei + E;

    // workspace carve-up (floats)
    float* ws = (float*)d_ws;
    const size_t NHC = (size_t)N * HC;
    float*    h1     = ws;                              // [N,256] gemm out / agg in
    float*    h2     = h1 + NHC;                        // [N,256] agg out / next gemm in
    float*    s1     = h2 + NHC;                        // [N,8]
    float*    s2     = s1 + (size_t)N * H;              // [N,8]
    unsigned* amax   = (unsigned*)(s2 + (size_t)N * H); // [N,8]
    float*    denom  = (float*)(amax + (size_t)N * H);  // [N,8]
    float*    ealpha = denom + (size_t)N * H;           // [E,8] alpha -> ex -> coef

    const unsigned EH   = (unsigned)E * H;
    const unsigned NH   = (unsigned)N * H;
    const unsigned NHCu = (unsigned)NHC;
    const unsigned EHC  = (unsigned)E * HC;             // 128M: fits u32

    for (int layer = 0; layer < 2; ++layer) {
        const float* gin = (layer == 0) ? x  : h2;
        const float* Wl  = (layer == 0) ? W1 : W2;
        const float* asl = (layer == 0) ? as1 : as2;
        const float* adl = (layer == 0) ? ad1 : ad2;
        const int    Kl  = (layer == 0) ? IN : HC;

        dim3 g(cdiv(N, 64), cdiv(HC, 128));
        k_gemm_wmma<<<g, 256, 0, stream>>>(gin, Wl, h1, N, Kl, HC);
        k_scores<8, 32><<<cdiv(NH, 256), 256, 0, stream>>>(h1, asl, adl, s1, s2, NH);
        k_fill_u32<<<cdiv(NH, 256), 256, 0, stream>>>(amax, KEY_NEG_INF, NH);
        k_fill_f32x4<<<cdiv(NH / 4, 256), 256, 0, stream>>>((float4*)denom, 0.f, NH / 4);
        k_fill_f32x4<<<cdiv(NHCu / 4, 256), 256, 0, stream>>>((float4*)h2, 0.f, NHCu / 4);
        k_edge_alpha<8><<<cdiv(EH, 256), 256, 0, stream>>>(src, dst, s1, s2, ealpha, amax, EH);
        k_edge_exp<8><<<cdiv(EH, 256), 256, 0, stream>>>(dst, ealpha, amax, denom, EH);
        k_edge_coef<8><<<cdiv(EH, 256), 256, 0, stream>>>(dst, ealpha, denom, EH);
        k_edge_agg<8, 32><<<cdiv(EHC, 256), 256, 0, stream>>>(src, dst, ealpha, h1, h2, EHC);
        k_elu4<<<cdiv(NHCu / 4, 256), 256, 0, stream>>>((float4*)h2, NHCu / 4);
    }

    // ---- Layer 3 (H=1, C=40; mean over single head == identity) + log_softmax
    {
        dim3 g(cdiv(N, 64), cdiv(OUT, 128));
        k_gemm_wmma<<<g, 256, 0, stream>>>(h2, W3, h1, N, HC, OUT);   // h1[:N*40]
        const unsigned Nu = (unsigned)N, Eu = (unsigned)E;
        const unsigned NOUT = (unsigned)N * OUT;
        const unsigned EC3  = (unsigned)E * OUT;                      // 20M
        k_scores<1, 40><<<cdiv(Nu, 256), 256, 0, stream>>>(h1, as3, ad3, s1, s2, Nu);
        k_fill_u32<<<cdiv(Nu, 256), 256, 0, stream>>>(amax, KEY_NEG_INF, Nu);
        k_fill_f32x4<<<cdiv(Nu / 4, 256), 256, 0, stream>>>((float4*)denom, 0.f, Nu / 4);
        k_fill_f32x4<<<cdiv(NOUT / 4, 256), 256, 0, stream>>>((float4*)h2, 0.f, NOUT / 4);
        k_edge_alpha<1><<<cdiv(Eu, 256), 256, 0, stream>>>(src, dst, s1, s2, ealpha, amax, Eu);
        k_edge_exp<1><<<cdiv(Eu, 256), 256, 0, stream>>>(dst, ealpha, amax, denom, Eu);
        k_edge_coef<1><<<cdiv(Eu, 256), 256, 0, stream>>>(dst, ealpha, denom, Eu);
        k_edge_agg<1, 40><<<cdiv(EC3, 256), 256, 0, stream>>>(src, dst, ealpha, h1, h2, EC3);
        k_logsoftmax<40><<<cdiv(Nu, 256), 256, 0, stream>>>(h2, (float*)d_out, Nu);
    }
}
